// CausalMultiHeadAttentionRoPE_3478923510050
// MI455X (gfx1250) — matrix operand order repaired and checked by reading, not compiled, not run on hardware
//
#include <hip/hip_runtime.h>

#define LSEQ   2048
#define DMODEL 1024
#define NHEADS 16
#define DKH    64
#define NTOK   4096   // B * L

typedef __attribute__((ext_vector_type(16))) __bf16 v16bf;
typedef __attribute__((ext_vector_type(8)))  float  v8f;
typedef __attribute__((ext_vector_type(4)))  unsigned int tdm_v4u;
typedef __attribute__((ext_vector_type(8)))  int          tdm_v8i;
typedef __attribute__((ext_vector_type(4)))  int          tdm_v4i;

#if __has_builtin(__builtin_amdgcn_tensor_load_to_lds) && \
    __has_builtin(__builtin_amdgcn_s_wait_tensorcnt)
#define TDM_HW 1
#define TDM_WAIT(n) __builtin_amdgcn_s_wait_tensorcnt((short)(n))
#else
#define TDM_WAIT(n)
#endif

union FragU { v16bf v; unsigned int u[8]; };

__device__ inline unsigned short bf16_of_f32(float x) {
  union { float f; unsigned int u; } c; c.f = x;
  unsigned int u = c.u;
  u += 0x7FFFu + ((u >> 16) & 1u);   // round-to-nearest-even
  return (unsigned short)(u >> 16);
}

// ---------------------------------------------------------------------------
// TDM: DMA one 64-wide bf16 tile (64 rows x 64 cols) global -> LDS.
// D# per ISA 8.3/8.4: count=1, type=2(image), data_size=2B,
// pad_enable with pad_interval=32 dwords (one 128B row) + pad_amount=1 dword
// => LDS row stride 66 shorts, matching the [64][66] conflict-free layout.
// tensor_dim == tile_dim (tile start is global_addr) => no OOB clamping.
// Issued by wave 0 only; TENSORcnt is per-wave and in-order.
// This toolchain's builtin is the 6-arg form (g0,g1,g2,g3,g4,cpol).
// ---------------------------------------------------------------------------
__device__ inline void tdm_tile64(unsigned short* lds, const unsigned short* g,
                                  int strideElems, int lane) {
#ifdef TDM_HW
  unsigned long long ga = (unsigned long long)(size_t)g;
  tdm_v4u g0;
  g0[0] = 1u;                                   // count=1, user mode
  g0[1] = (unsigned)(size_t)lds;                // flat low32 == LDS byte offset
  g0[2] = (unsigned)ga;                         // global_addr[31:0]
  g0[3] = (unsigned)(ga >> 32) | (2u << 30);    // global_addr[56:32] | type=2
  tdm_v8i g1;
  g1[0] = (1 << 16)        // data_size = 2 bytes
        | (1 << 20)        // pad_enable
        | (4 << 22);       // pad_interval: 2^(4+1)=32 dwords, pad_amount=1 dword
  g1[1] = (int)(64u << 16);            // tensor_dim0[15:0]=64
  g1[2] = (int)(64u << 16);            // tensor_dim0 hi=0 | tensor_dim1[15:0]=64
  g1[3] = (int)(64u << 16);            // tensor_dim1 hi=0 | tile_dim0=64
  g1[4] = 64;                          // tile_dim1=64, tile_dim2=0
  g1[5] = strideElems;                 // tensor_dim0_stride[31:0]
  g1[6] = 0;
  g1[7] = 0;
  tdm_v4i z4 = {0, 0, 0, 0};
  tdm_v8i z8 = {0, 0, 0, 0, 0, 0, 0, 0};
  __builtin_amdgcn_tensor_load_to_lds(g0, g1, z4, z4, z8, 0);
#else
  for (int i = lane; i < 64 * 64; i += 32) {
    int r = i >> 6, c = i & 63;
    lds[r * 66 + c] = g[(size_t)r * strideElems + c];
  }
#endif
}

// A fragment (16x32 bf16, row-major src: src[m*ld + k]).
__device__ inline v16bf load_frag_a(const unsigned short* src, int ld, int lane) {
  FragU f;
  const unsigned short* p = src + (lane & 15) * ld + ((lane >> 4) << 4);
#pragma unroll
  for (int v = 0; v < 8; ++v) {
    int k = v << 1;   // 2*v
    f.u[v] = *(const unsigned int*)(p + k);
  }
  return f.v;
}

// B fragment (KxN) where src holds B transposed: src[n*ld + k].
__device__ inline v16bf load_frag_b(const unsigned short* src, int ld, int lane) {
  FragU f;
  const unsigned short* p = src + (lane & 15) * ld + ((lane >> 4) << 4);
#pragma unroll
  for (int v = 0; v < 8; ++v)
    f.u[v] = *(const unsigned int*)(p + (v << 1));
  return f.v;
}

// B fragment where src is row-major K x N: src[k*ld + n] (paired 16-bit loads).
__device__ inline v16bf load_frag_b_rm(const unsigned short* src, int ld, int lane) {
  FragU f;
  const unsigned short* p = src + (lane & 15) + ((lane >> 4) << 4) * ld;
#pragma unroll
  for (int v = 0; v < 8; ++v) {
    unsigned lo = p[(2 * v) * ld];
    unsigned hi = p[(2 * v + 1) * ld];
    f.u[v] = lo | (hi << 16);
  }
  return f.v;
}

__device__ inline v8f wmma_bf16(v16bf a, v16bf b, v8f c) {
  return __builtin_amdgcn_wmma_f32_16x16x32_bf16(false, a, false, b, (short)0, c,
                                                 false, false);
}

// ---------------------------------------------------------------------------
// Kernel 0: one-shot f32 -> bf16 conversion (X, Wqkv, Wout).
// ---------------------------------------------------------------------------
__global__ __launch_bounds__(256) void cvt_f32_bf16_kernel(
    const float* __restrict__ s, unsigned short* __restrict__ d, int n4) {
  int i = blockIdx.x * 256 + threadIdx.x;
  if (i < n4) {
    float4 f = ((const float4*)s)[i];
    unsigned lo = (unsigned)bf16_of_f32(f.x) | ((unsigned)bf16_of_f32(f.y) << 16);
    unsigned hi = (unsigned)bf16_of_f32(f.z) | ((unsigned)bf16_of_f32(f.w) << 16);
    ((uint2*)d)[i] = make_uint2(lo, hi);
  }
}

// ---------------------------------------------------------------------------
// Kernel 1: QKV = X @ Wqkv^T (bf16 WMMA, f32 acc), TDM-fed + double-buffered,
// fused RoPE on Q,K. Grid (64,48), block 128 (4 waves, 32x32 subtiles).
// ---------------------------------------------------------------------------
__global__ __launch_bounds__(128) void qkv_rope_kernel(
    const unsigned short* __restrict__ Xb, const int* __restrict__ tpos,
    const unsigned short* __restrict__ Wqb,
    unsigned short* __restrict__ Qw, unsigned short* __restrict__ Kw,
    unsigned short* __restrict__ Vw) {
  __shared__ unsigned short As[2][64][66];
  __shared__ unsigned short Bs[2][64][66];
  const int tid = threadIdx.x;
  const int lane = tid & 31, wave = tid >> 5;
  const int wm = wave >> 1, wn = wave & 1;
  const int mBase = blockIdx.x << 6;
  const int nBase = blockIdx.y << 6;
  v8f acc00 = {}, acc01 = {}, acc10 = {}, acc11 = {};

  if (wave == 0) {
    tdm_tile64(&As[0][0][0], Xb + (size_t)mBase * DMODEL, DMODEL, lane);
    tdm_tile64(&Bs[0][0][0], Wqb + (size_t)nBase * DMODEL, DMODEL, lane);
  }
  for (int it = 0; it < DMODEL / 64; ++it) {
    const int buf = it & 1;
    if (wave == 0) {
      if (it + 1 < DMODEL / 64) {  // prefetch next K-chunk into other buffer
        tdm_tile64(&As[buf ^ 1][0][0],
                   Xb + (size_t)mBase * DMODEL + (it + 1) * 64, DMODEL, lane);
        tdm_tile64(&Bs[buf ^ 1][0][0],
                   Wqb + (size_t)nBase * DMODEL + (it + 1) * 64, DMODEL, lane);
        TDM_WAIT(2);   // oldest 2 (this chunk's A,B) complete
      } else {
        TDM_WAIT(0);
      }
    }
    __syncthreads();
#pragma unroll
    for (int kk = 0; kk < 64; kk += 32) {
      v16bf a0 = load_frag_a(&As[buf][wm * 32][kk],      66, lane);
      v16bf a1 = load_frag_a(&As[buf][wm * 32 + 16][kk], 66, lane);
      v16bf b0 = load_frag_b(&Bs[buf][wn * 32][kk],      66, lane);
      v16bf b1 = load_frag_b(&Bs[buf][wn * 32 + 16][kk], 66, lane);
      acc00 = wmma_bf16(a0, b0, acc00);
      acc01 = wmma_bf16(a0, b1, acc01);
      acc10 = wmma_bf16(a1, b0, acc10);
      acc11 = wmma_bf16(a1, b1, acc11);
    }
    __syncthreads();  // all reads of buf done before it is DMA-overwritten
  }

  const int rowHalf = (lane >> 4) << 3;  // C layout: M = r + 8*(lane>=16)
  const int colL = lane & 15;            //           N = lane & 15
  v8f accs[2][2] = {{acc00, acc01}, {acc10, acc11}};
#pragma unroll
  for (int sm = 0; sm < 2; ++sm)
#pragma unroll
    for (int sn = 0; sn < 2; ++sn) {
      v8f c = accs[sm][sn];
      const int fB = nBase + wn * 32 + sn * 16 + colL;  // feature in [0,3072)
#pragma unroll
      for (int r = 0; r < 8; ++r) {
        int t = mBase + wm * 32 + sm * 16 + rowHalf + r;  // token in [0,4096)
        float v = c[r];
        int b = t >> 11, l = t & (LSEQ - 1);
        if (fB < 2 * DMODEL) {  // Q or K: RoPE (branch is wave-uniform)
          float partner = __shfl_xor(v, 1, 32);  // paired even/odd feature
          int dk = fB & 63;
          float p2 = (float)(dk >> 1);
          float inv = __expf(p2 * (-2.0f / 64.0f) * 9.2103403719761836f);
          float ang = (float)tpos[t] * inv;
          float s, cz;
          __sincosf(ang, &s, &cz);
          float outv = (fB & 1) ? (v * cz + partner * s)
                                : (v * cz - partner * s);
          int h = (fB & (DMODEL - 1)) >> 6;
          unsigned short* dst = (fB < DMODEL) ? Qw : Kw;
          dst[((((size_t)b * NHEADS + h) * LSEQ) + l) * DKH + dk] =
              bf16_of_f32(outv);
        } else {                // V: passthrough
          int fv = fB - 2 * DMODEL;
          int h = fv >> 6, dk = fv & 63;
          Vw[((((size_t)b * NHEADS + h) * LSEQ) + l) * DKH + dk] =
              bf16_of_f32(v);
        }
      }
    }
}

// ---------------------------------------------------------------------------
// Kernel 2: causal flash attention per (b,h), 64 queries/block, TDM-fed K/V
// with double buffering. Grid (32,16,2), block 128.
// ---------------------------------------------------------------------------
__global__ __launch_bounds__(128) void attn_kernel(
    const unsigned short* __restrict__ Qw, const unsigned short* __restrict__ Kw,
    const unsigned short* __restrict__ Vw, unsigned short* __restrict__ Ow) {
  __shared__ unsigned short Qs[64][66];
  __shared__ unsigned short Ks[2][64][66];
  __shared__ unsigned short Vs[2][64][66];   // row-major (key, dk)
  __shared__ unsigned short Ps[64][66];      // softmax weights, wave-private rows
  const int tid = threadIdx.x, lane = tid & 31, wave = tid >> 5;
  const int qTile = blockIdx.x;
  const int h = blockIdx.y, b = blockIdx.z;
  const size_t headBase = ((size_t)b * NHEADS + h) * LSEQ * DKH;
  const unsigned short* Qh = Qw + headBase;
  const unsigned short* Kh = Kw + headBase;
  const unsigned short* Vh = Vw + headBase;
  const int qBase = qTile << 6;
  const int rowHalf = (lane >> 4) << 3, colL = lane & 15;

  if (wave == 0) {
    tdm_tile64(&Qs[0][0],    Qh + (size_t)qBase * DKH, DKH, lane);
    tdm_tile64(&Ks[0][0][0], Kh, DKH, lane);
    tdm_tile64(&Vs[0][0][0], Vh, DKH, lane);
    TDM_WAIT(2);  // Q (oldest) complete
  }
  __syncthreads();
  v16bf qa0 = load_frag_a(&Qs[wave * 16][0],  66, lane);
  v16bf qa1 = load_frag_a(&Qs[wave * 16][32], 66, lane);

  v8f o[4] = {};
  float mi[8], li[8];
#pragma unroll
  for (int r = 0; r < 8; ++r) { mi[r] = -__builtin_inff(); li[r] = 0.f; }

  for (int j = 0; j <= qTile; ++j) {  // causal: skip fully masked key blocks
    const int buf = j & 1;
    if (wave == 0) {
      if (j < qTile) {   // prefetch next K/V block
        tdm_tile64(&Ks[buf ^ 1][0][0], Kh + (size_t)(j + 1) * 64 * DKH, DKH, lane);
        tdm_tile64(&Vs[buf ^ 1][0][0], Vh + (size_t)(j + 1) * 64 * DKH, DKH, lane);
        TDM_WAIT(2);   // this block's K,V complete
      } else {
        TDM_WAIT(0);
      }
    }
    __syncthreads();

    // S = Q K^T  (Ks row-major (key,dk) == transposed-B layout)
    v8f s[4];
#pragma unroll
    for (int sn = 0; sn < 4; ++sn) {
      v8f a = {};
      a = wmma_bf16(qa0, load_frag_b(&Ks[buf][sn * 16][0],  66, lane), a);
      a = wmma_bf16(qa1, load_frag_b(&Ks[buf][sn * 16][32], 66, lane), a);
      s[sn] = a;
    }

    // scale + causal mask (only diagonal block needs masking)
#pragma unroll
    for (int sn = 0; sn < 4; ++sn)
#pragma unroll
      for (int r = 0; r < 8; ++r) {
        float x = s[sn][r] * 0.125f;  // 1/sqrt(64)
        if (j == qTile) {
          int kg = (j << 6) + sn * 16 + colL;
          int qg = qBase + wave * 16 + rowHalf + r;
          if (kg > qg) x = -__builtin_inff();
        }
        s[sn][r] = x;
      }

    // online softmax (row = 16-lane group in C layout)
#pragma unroll
    for (int r = 0; r < 8; ++r) {
      float mb = fmaxf(fmaxf(s[0][r], s[1][r]), fmaxf(s[2][r], s[3][r]));
      mb = fmaxf(mb, __shfl_xor(mb, 1, 32));
      mb = fmaxf(mb, __shfl_xor(mb, 2, 32));
      mb = fmaxf(mb, __shfl_xor(mb, 4, 32));
      mb = fmaxf(mb, __shfl_xor(mb, 8, 32));
      float mnew = fmaxf(mi[r], mb);
      float sc = __expf(mi[r] - mnew);
      float psum = 0.f;
#pragma unroll
      for (int sn = 0; sn < 4; ++sn) {
        float p = __expf(s[sn][r] - mnew);
        s[sn][r] = p;
        psum += p;
      }
      psum += __shfl_xor(psum, 1, 32);
      psum += __shfl_xor(psum, 2, 32);
      psum += __shfl_xor(psum, 4, 32);
      psum += __shfl_xor(psum, 8, 32);
      li[r] = li[r] * sc + psum;
      mi[r] = mnew;
#pragma unroll
      for (int sn = 0; sn < 4; ++sn) o[sn][r] *= sc;
    }

    // stage P through LDS: C-fragment -> A-fragment (wave-private rows)
#pragma unroll
    for (int sn = 0; sn < 4; ++sn)
#pragma unroll
      for (int r = 0; r < 8; ++r)
        Ps[wave * 16 + rowHalf + r][sn * 16 + colL] = bf16_of_f32(s[sn][r]);

    v16bf pa0 = load_frag_a(&Ps[wave * 16][0],  66, lane);
    v16bf pa1 = load_frag_a(&Ps[wave * 16][32], 66, lane);
#pragma unroll
    for (int sn = 0; sn < 4; ++sn) {
      o[sn] = wmma_bf16(pa0, load_frag_b_rm(&Vs[buf][0][sn * 16],  66, lane), o[sn]);
      o[sn] = wmma_bf16(pa1, load_frag_b_rm(&Vs[buf][32][sn * 16], 66, lane), o[sn]);
    }
    __syncthreads();  // reads of buf done before it is DMA-overwritten
  }

  // normalize, write O in (token, d_model) bf16
#pragma unroll
  for (int r = 0; r < 8; ++r) {
    float inv = 1.0f / li[r];
    int q = qBase + wave * 16 + rowHalf + r;
    size_t t = (size_t)b * LSEQ + q;
#pragma unroll
    for (int sn = 0; sn < 4; ++sn) {
      int dk = sn * 16 + colL;
      Ow[t * DMODEL + h * DKH + dk] = bf16_of_f32(o[sn][r] * inv);
    }
  }
}

// ---------------------------------------------------------------------------
// Kernel 3: Y = O @ Wout^T (bf16 WMMA, f32 out), TDM-fed + double-buffered.
// Grid (64,16), block 128.
// ---------------------------------------------------------------------------
__global__ __launch_bounds__(128) void out_proj_kernel(
    const unsigned short* __restrict__ Ow, const unsigned short* __restrict__ Wob,
    float* __restrict__ Y) {
  __shared__ unsigned short As[2][64][66];
  __shared__ unsigned short Bs[2][64][66];
  const int tid = threadIdx.x, lane = tid & 31, wave = tid >> 5;
  const int wm = wave >> 1, wn = wave & 1;
  const int mBase = blockIdx.x << 6, nBase = blockIdx.y << 6;
  v8f acc00 = {}, acc01 = {}, acc10 = {}, acc11 = {};

  if (wave == 0) {
    tdm_tile64(&As[0][0][0], Ow  + (size_t)mBase * DMODEL, DMODEL, lane);
    tdm_tile64(&Bs[0][0][0], Wob + (size_t)nBase * DMODEL, DMODEL, lane);
  }
  for (int it = 0; it < DMODEL / 64; ++it) {
    const int buf = it & 1;
    if (wave == 0) {
      if (it + 1 < DMODEL / 64) {
        tdm_tile64(&As[buf ^ 1][0][0],
                   Ow + (size_t)mBase * DMODEL + (it + 1) * 64, DMODEL, lane);
        tdm_tile64(&Bs[buf ^ 1][0][0],
                   Wob + (size_t)nBase * DMODEL + (it + 1) * 64, DMODEL, lane);
        TDM_WAIT(2);
      } else {
        TDM_WAIT(0);
      }
    }
    __syncthreads();
#pragma unroll
    for (int kk = 0; kk < 64; kk += 32) {
      v16bf a0 = load_frag_a(&As[buf][wm * 32][kk],      66, lane);
      v16bf a1 = load_frag_a(&As[buf][wm * 32 + 16][kk], 66, lane);
      v16bf b0 = load_frag_b(&Bs[buf][wn * 32][kk],      66, lane);
      v16bf b1 = load_frag_b(&Bs[buf][wn * 32 + 16][kk], 66, lane);
      acc00 = wmma_bf16(a0, b0, acc00);
      acc01 = wmma_bf16(a0, b1, acc01);
      acc10 = wmma_bf16(a1, b0, acc10);
      acc11 = wmma_bf16(a1, b1, acc11);
    }
    __syncthreads();
  }

  const int rowHalf = (lane >> 4) << 3, colL = lane & 15;
  v8f accs[2][2] = {{acc00, acc01}, {acc10, acc11}};
#pragma unroll
  for (int sm = 0; sm < 2; ++sm)
#pragma unroll
    for (int sn = 0; sn < 2; ++sn) {
      v8f c = accs[sm][sn];
#pragma unroll
      for (int r = 0; r < 8; ++r) {
        int t = mBase + wm * 32 + sm * 16 + rowHalf + r;
        int f = nBase + wn * 32 + sn * 16 + colL;
        Y[(size_t)t * DMODEL + f] = c[r];
      }
    }
}

// ---------------------------------------------------------------------------
extern "C" void kernel_launch(void* const* d_in, const int* in_sizes, int n_in,
                              void* d_out, int out_size, void* d_ws, size_t ws_size,
                              hipStream_t stream) {
  (void)in_sizes; (void)n_in; (void)out_size; (void)ws_size;
  const float* X    = (const float*)d_in[0];
  const int*   tpos = (const int*)d_in[1];
  const float* Wqkv = (const float*)d_in[2];
  const float* Wout = (const float*)d_in[3];
  float* Y = (float*)d_out;

  // workspace layout (bf16 elements)
  unsigned short* Xb  = (unsigned short*)d_ws;                 // 4M
  unsigned short* Wqb = Xb  + (size_t)NTOK * DMODEL;           // 3M
  unsigned short* Wob = Wqb + (size_t)3 * DMODEL * DMODEL;     // 1M
  unsigned short* Qw  = Wob + (size_t)DMODEL * DMODEL;         // 4M
  unsigned short* Kw  = Qw  + (size_t)NTOK * DMODEL;           // 4M
  unsigned short* Vw  = Kw  + (size_t)NTOK * DMODEL;           // 4M
  unsigned short* Ow  = Vw  + (size_t)NTOK * DMODEL;           // 4M

  // one-shot f32 -> bf16 conversion
  cvt_f32_bf16_kernel<<<(NTOK * DMODEL / 4 + 255) / 256, 256, 0, stream>>>(
      X, Xb, NTOK * DMODEL / 4);
  cvt_f32_bf16_kernel<<<(3 * DMODEL * DMODEL / 4 + 255) / 256, 256, 0, stream>>>(
      Wqkv, Wqb, 3 * DMODEL * DMODEL / 4);
  cvt_f32_bf16_kernel<<<(DMODEL * DMODEL / 4 + 255) / 256, 256, 0, stream>>>(
      Wout, Wob, DMODEL * DMODEL / 4);

  dim3 g1(NTOK / 64, 3 * DMODEL / 64);  // (64, 48)
  qkv_rope_kernel<<<g1, 128, 0, stream>>>(Xb, tpos, Wqb, Qw, Kw, Vw);

  dim3 g2(LSEQ / 64, NHEADS, 2);        // (32, 16, 2)
  attn_kernel<<<g2, 128, 0, stream>>>(Qw, Kw, Vw, Ow);

  dim3 g3(NTOK / 64, DMODEL / 64);      // (64, 16)
  out_proj_kernel<<<g3, 128, 0, stream>>>(Ow, Wob, Y);
}
